// XdimResBlock_77618648973582
// MI455X (gfx1250) — matrix operand order repaired and checked by reading, not compiled, not run on hardware
//
#include <hip/hip_runtime.h>
#include <hip/hip_bf16.h>
#include <math.h>

// ---------------- problem constants ----------------
static constexpr int B = 4, T = 20000, N = 60000, H = 128, V = 128;

// ---------------- WMMA types / helpers ----------------
typedef __bf16 bf16_t;
typedef __attribute__((ext_vector_type(16))) __bf16 v16bf;
typedef __attribute__((ext_vector_type(8)))  float  v8f;

// Build a 16-element bf16 fragment from two 8-element (16B aligned) runs.
// Matches CDNA5 16-bit A/B fragment layouts (two K-octets per lane).
static __device__ __forceinline__ v16bf mkfrag(const bf16_t* __restrict__ p0,
                                               const bf16_t* __restrict__ p1) {
  v16bf f;
#pragma unroll
  for (int i = 0; i < 8; ++i) { f[i] = p0[i]; f[8 + i] = p1[i]; }
  return f;
}

static __device__ __forceinline__ v8f wmma_bf16(v16bf a, v16bf b, v8f c) {
  // (neg_a, A, neg_b, B, c_mod, C, reuse_a, reuse_b)
  return __builtin_amdgcn_wmma_f32_16x16x32_bf16(false, a, false, b, (short)0, c,
                                                 false, false);
}

static __device__ __forceinline__ float gelu_erf(float x) {
  return 0.5f * x * (1.0f + erff(x * 0.70710678118654752f));
}

// ---------------- weight prep: f32 (K,Nout) -> bf16 transposed (Nout,K) ---------
__global__ void k_prep(const float* __restrict__ W, bf16_t* __restrict__ Wt,
                       int K, int Nout) {
  int i = blockIdx.x * blockDim.x + threadIdx.x;
  if (i >= K * Nout) return;
  int n = i / K, k = i - n * K;
  Wt[i] = (bf16_t)W[k * Nout + n];
}

// ---------------- K1: inflate (gather 3 hexes, K=384 GEMM, +residual) ----------
// 32 output rows per workgroup; each wave owns one 16-col tile and reuses every
// B-fragment across two M-tiles (2 WMMAs per B-load batch).
__global__ __launch_bounds__(256) void k_inflate(
    const float* __restrict__ hexf, const float* __restrict__ vertf,
    const bf16_t* __restrict__ Winf, const float* __restrict__ binf,
    const int* __restrict__ vth, float* __restrict__ v1) {
  __shared__ __align__(16) bf16_t A[32 * 392];  // 32 rows x 384 (pad->392)
  const int tid = threadIdx.x;
  const long g0 = (long)blockIdx.x * 32;        // flat (b*N + n) row base
  const int b  = (int)(g0 / N);
  const int n0 = (int)(g0 % N);

  for (int e = tid; e < 32 * 384; e += 256) {
    int r = e / 384, k = e - r * 384;
    int j = k >> 7, c = k & 127;
    int idx = vth[(n0 + r) * 3 + j];
    float val = (idx >= 0) ? hexf[((long)b * T + idx) * H + c] : 0.0f;
    A[r * 392 + k] = (bf16_t)val;
  }
  __syncthreads();

  const int lane = tid & 31, wv = tid >> 5;
  const int half = lane >> 4, l16 = lane & 15;
  const int col = wv * 16 + l16;
  v8f acc0 = {}, acc1 = {};
#pragma unroll
  for (int kc = 0; kc < 384; kc += 32) {
    const bf16_t* b0 = &Winf[(long)col * 384 + kc + half * 8];
    v16bf bf_ = mkfrag(b0, b0 + 16);
    const bf16_t* a0 = &A[l16 * 392 + kc + half * 8];
    const bf16_t* a1 = &A[(16 + l16) * 392 + kc + half * 8];
    acc0 = wmma_bf16(mkfrag(a0, a0 + 16), bf_, acc0);
    acc1 = wmma_bf16(mkfrag(a1, a1 + 16), bf_, acc1);
  }
  const float bias = binf[col];
#pragma unroll
  for (int r = 0; r < 8; ++r) {
    long row0 = g0 + r + 8 * half;
    long o0 = row0 * V + col;
    v1[o0] = acc0[r] + bias + vertf[o0];
    long o1 = (row0 + 16) * V + col;
    v1[o1] = acc1[r] + bias + vertf[o1];
  }
}

// ---------------- K2: message pass + update + LayerNorm -------------------------
__global__ __launch_bounds__(256) void k_msg(
    const float* __restrict__ v1, const bf16_t* __restrict__ Wmsg,
    const float* __restrict__ bmsg, const bf16_t* __restrict__ Wupd,
    const float* __restrict__ bupd, const float* __restrict__ vng,
    const float* __restrict__ vnb, const int* __restrict__ vadj,
    float* __restrict__ v2out) {
  __shared__ __align__(16) bf16_t Av[32 * 136];  // own rows (bf16)
  __shared__ __align__(16) bf16_t As[32 * 136];  // neighbor sums (bf16)
  __shared__ __align__(16) bf16_t Ag[32 * 136];  // agg rows (bf16)
  __shared__ float Xf[32 * 132];                 // own rows f32 (residual)
  __shared__ float Ct[32 * 132];                 // pre-LN result
  __shared__ float invc[32];
  const int tid = threadIdx.x;
  const long g0 = (long)blockIdx.x * 32;
  const int b  = (int)(g0 / N);
  const int n0 = (int)(g0 % N);

  for (int e = tid; e < 32 * 128; e += 256) {
    int r = e >> 7, c = e & 127;
    float v = v1[(g0 + r) * V + c];
    Av[r * 136 + c] = (bf16_t)v;
    Xf[r * 132 + c] = v;
    float s = 0.0f;
#pragma unroll
    for (int j = 0; j < 3; ++j) {
      int idx = vadj[(n0 + r) * 3 + j];
      if (idx >= 0) s += v1[((long)b * N + idx) * V + c];
    }
    As[r * 136 + c] = (bf16_t)s;
  }
  if (tid < 32) {
    int cnt = 0;
#pragma unroll
    for (int j = 0; j < 3; ++j) cnt += (vadj[(n0 + tid) * 3 + j] >= 0);
    invc[tid] = 1.0f / fmaxf((float)cnt, 1.0f);
  }
  __syncthreads();

  const int lane = tid & 31, wv = tid >> 5;
  const int half = lane >> 4, l16 = lane & 15;
  const int col = wv * 16 + l16;

  // GEMM1: Hmsg = S @ Wmsg   (agg = (Hmsg + 3*b_msg) / cnt)
  {
    v8f acc0 = {}, acc1 = {};
#pragma unroll
    for (int kc = 0; kc < 128; kc += 32) {
      const bf16_t* b0 = &Wmsg[(long)col * 128 + kc + half * 8];
      v16bf bf_ = mkfrag(b0, b0 + 16);
      const bf16_t* a0 = &As[l16 * 136 + kc + half * 8];
      const bf16_t* a1 = &As[(16 + l16) * 136 + kc + half * 8];
      acc0 = wmma_bf16(mkfrag(a0, a0 + 16), bf_, acc0);
      acc1 = wmma_bf16(mkfrag(a1, a1 + 16), bf_, acc1);
    }
    const float bm3 = bmsg[col] * 3.0f;
#pragma unroll
    for (int r = 0; r < 8; ++r) {
      int row = r + 8 * half;
      Ag[row * 136 + col] = (bf16_t)((acc0[r] + bm3) * invc[row]);
      Ag[(row + 16) * 136 + col] = (bf16_t)((acc1[r] + bm3) * invc[row + 16]);
    }
  }
  __syncthreads();

  // GEMM2: mp = v1 @ Wupd[:,0:128] + agg @ Wupd[:,128:256]
  {
    v8f acc0 = {}, acc1 = {};
#pragma unroll
    for (int kc = 0; kc < 128; kc += 32) {
      const bf16_t* b0 = &Wupd[(long)col * 256 + kc + half * 8];
      v16bf bf_ = mkfrag(b0, b0 + 16);
      const bf16_t* a0 = &Av[l16 * 136 + kc + half * 8];
      const bf16_t* a1 = &Av[(16 + l16) * 136 + kc + half * 8];
      acc0 = wmma_bf16(mkfrag(a0, a0 + 16), bf_, acc0);
      acc1 = wmma_bf16(mkfrag(a1, a1 + 16), bf_, acc1);
    }
#pragma unroll
    for (int kc = 0; kc < 128; kc += 32) {
      const bf16_t* b0 = &Wupd[(long)col * 256 + 128 + kc + half * 8];
      v16bf bf_ = mkfrag(b0, b0 + 16);
      const bf16_t* a0 = &Ag[l16 * 136 + kc + half * 8];
      const bf16_t* a1 = &Ag[(16 + l16) * 136 + kc + half * 8];
      acc0 = wmma_bf16(mkfrag(a0, a0 + 16), bf_, acc0);
      acc1 = wmma_bf16(mkfrag(a1, a1 + 16), bf_, acc1);
    }
    const float bu = bupd[col];
#pragma unroll
    for (int r = 0; r < 8; ++r) {
      int row = r + 8 * half;
      Ct[row * 132 + col] = Xf[row * 132 + col] + acc0[r] + bu;
      Ct[(row + 16) * 132 + col] = Xf[(row + 16) * 132 + col] + acc1[r] + bu;
    }
  }
  __syncthreads();

  // LayerNorm: wave wv handles rows 4wv..4wv+3; 32 lanes x 4 cols each.
#pragma unroll
  for (int rr = 0; rr < 4; ++rr) {
    int row = wv * 4 + rr;
    float x[4], s0 = 0.0f, s1 = 0.0f;
#pragma unroll
    for (int q = 0; q < 4; ++q) {
      x[q] = Ct[row * 132 + lane * 4 + q];
      s0 += x[q]; s1 += x[q] * x[q];
    }
#pragma unroll
    for (int off = 16; off >= 1; off >>= 1) {
      s0 += __shfl_xor(s0, off, 32);
      s1 += __shfl_xor(s1, off, 32);
    }
    float mu = s0 * (1.0f / 128.0f);
    float var = s1 * (1.0f / 128.0f) - mu * mu;
    float rs = rsqrtf(var + 1e-5f);
#pragma unroll
    for (int q = 0; q < 4; ++q) {
      int c = lane * 4 + q;
      v2out[(g0 + row) * V + c] = (x[q] - mu) * rs * vng[c] + vnb[c];
    }
  }
}

// ---------------- K3/K5: residual MLP 128 -> 256 (erf-GELU) -> 128, in place ----
__global__ __launch_bounds__(256) void k_mlp(
    float* __restrict__ io, const bf16_t* __restrict__ W1,
    const float* __restrict__ b1, const bf16_t* __restrict__ W2,
    const float* __restrict__ b2) {
  __shared__ __align__(16) bf16_t A[32 * 136];
  __shared__ float Xf[32 * 132];
  __shared__ __align__(16) bf16_t Hh[32 * 264];  // 32 x 256 gelu hidden (pad->264)
  const int tid = threadIdx.x;
  const long g0 = (long)blockIdx.x * 32;

  for (int e = tid; e < 32 * 128; e += 256) {
    int r = e >> 7, c = e & 127;
    float v = io[(g0 + r) * 128 + c];
    A[r * 136 + c] = (bf16_t)v;
    Xf[r * 132 + c] = v;
  }
  __syncthreads();

  const int lane = tid & 31, wv = tid >> 5;
  const int half = lane >> 4, l16 = lane & 15;

  // GEMM1: 256 outputs -> each wave owns two 16-col tiles x two 16-row tiles.
#pragma unroll
  for (int t = 0; t < 2; ++t) {
    int col = (wv + 8 * t) * 16 + l16;
    v8f acc0 = {}, acc1 = {};
#pragma unroll
    for (int kc = 0; kc < 128; kc += 32) {
      const bf16_t* b0 = &W1[(long)col * 128 + kc + half * 8];
      v16bf bf_ = mkfrag(b0, b0 + 16);
      const bf16_t* a0 = &A[l16 * 136 + kc + half * 8];
      const bf16_t* a1 = &A[(16 + l16) * 136 + kc + half * 8];
      acc0 = wmma_bf16(mkfrag(a0, a0 + 16), bf_, acc0);
      acc1 = wmma_bf16(mkfrag(a1, a1 + 16), bf_, acc1);
    }
    float bb = b1[col];
#pragma unroll
    for (int r = 0; r < 8; ++r) {
      int row = r + 8 * half;
      Hh[row * 264 + col] = (bf16_t)gelu_erf(acc0[r] + bb);
      Hh[(row + 16) * 264 + col] = (bf16_t)gelu_erf(acc1[r] + bb);
    }
  }
  __syncthreads();

  // GEMM2: K=256 back to 128, + bias + residual, stored in place.
  const int col = wv * 16 + l16;
  v8f acc0 = {}, acc1 = {};
#pragma unroll
  for (int kc = 0; kc < 256; kc += 32) {
    const bf16_t* b0 = &W2[(long)col * 256 + kc + half * 8];
    v16bf bf_ = mkfrag(b0, b0 + 16);
    const bf16_t* a0 = &Hh[l16 * 264 + kc + half * 8];
    const bf16_t* a1 = &Hh[(16 + l16) * 264 + kc + half * 8];
    acc0 = wmma_bf16(mkfrag(a0, a0 + 16), bf_, acc0);
    acc1 = wmma_bf16(mkfrag(a1, a1 + 16), bf_, acc1);
  }
  const float bb = b2[col];
#pragma unroll
  for (int r = 0; r < 8; ++r) {
    int row = r + 8 * half;
    io[(g0 + row) * 128 + col] = Xf[row * 132 + col] + acc0[r] + bb;
    io[(g0 + row + 16) * 128 + col] = Xf[(row + 16) * 132 + col] + acc1[r] + bb;
  }
}

// ---------------- K4: deflate (gather 6 vertices, GEMM, residual, LayerNorm) ----
__global__ __launch_bounds__(256) void k_deflate(
    const float* __restrict__ vfin, const float* __restrict__ hexf,
    const bf16_t* __restrict__ Wdef, const float* __restrict__ bdef,
    const float* __restrict__ hng, const float* __restrict__ hnb,
    const int* __restrict__ htv, float* __restrict__ hexout) {
  __shared__ __align__(16) bf16_t A[32 * 136];  // pooled*cnt (scale post-GEMM)
  __shared__ float Ct[32 * 132];
  __shared__ float invc[32];
  const int tid = threadIdx.x;
  const long g0 = (long)blockIdx.x * 32;  // flat (b*T + t) rows
  const int b  = (int)(g0 / T);
  const int t0 = (int)(g0 % T);

  for (int e = tid; e < 32 * 128; e += 256) {
    int r = e >> 7, c = e & 127;
    float s = 0.0f;
#pragma unroll
    for (int j = 0; j < 6; ++j) {
      int idx = htv[(t0 + r) * 6 + j];
      if (idx >= 0) s += vfin[((long)b * N + idx) * V + c];
    }
    A[r * 136 + c] = (bf16_t)s;
  }
  if (tid < 32) {
    int cnt = 0;
#pragma unroll
    for (int j = 0; j < 6; ++j) cnt += (htv[(t0 + tid) * 6 + j] >= 0);
    invc[tid] = 1.0f / fmaxf((float)cnt, 1.0f);
  }
  __syncthreads();

  const int lane = tid & 31, wv = tid >> 5;
  const int half = lane >> 4, l16 = lane & 15;
  const int col = wv * 16 + l16;
  v8f acc0 = {}, acc1 = {};
#pragma unroll
  for (int kc = 0; kc < 128; kc += 32) {
    const bf16_t* b0 = &Wdef[(long)col * 128 + kc + half * 8];
    v16bf bf_ = mkfrag(b0, b0 + 16);
    const bf16_t* a0 = &A[l16 * 136 + kc + half * 8];
    const bf16_t* a1 = &A[(16 + l16) * 136 + kc + half * 8];
    acc0 = wmma_bf16(mkfrag(a0, a0 + 16), bf_, acc0);
    acc1 = wmma_bf16(mkfrag(a1, a1 + 16), bf_, acc1);
  }
  const float bd = bdef[col];
#pragma unroll
  for (int r = 0; r < 8; ++r) {
    int row = r + 8 * half;
    Ct[row * 132 + col] =
        hexf[(g0 + row) * H + col] + acc0[r] * invc[row] + bd;
    Ct[(row + 16) * 132 + col] =
        hexf[(g0 + row + 16) * H + col] + acc1[r] * invc[row + 16] + bd;
  }
  __syncthreads();

#pragma unroll
  for (int rr = 0; rr < 4; ++rr) {
    int row = wv * 4 + rr;
    float x[4], s0 = 0.0f, s1 = 0.0f;
#pragma unroll
    for (int q = 0; q < 4; ++q) {
      x[q] = Ct[row * 132 + lane * 4 + q];
      s0 += x[q]; s1 += x[q] * x[q];
    }
#pragma unroll
    for (int off = 16; off >= 1; off >>= 1) {
      s0 += __shfl_xor(s0, off, 32);
      s1 += __shfl_xor(s1, off, 32);
    }
    float mu = s0 * (1.0f / 128.0f);
    float var = s1 * (1.0f / 128.0f) - mu * mu;
    float rs = rsqrtf(var + 1e-5f);
#pragma unroll
    for (int q = 0; q < 4; ++q) {
      int c = lane * 4 + q;
      hexout[(g0 + row) * H + c] = (x[q] - mu) * rs * hng[c] + hnb[c];
    }
  }
}

// ---------------- workspace layout (bf16 element offsets) -----------------------
static constexpr size_t OFF_INF = 0;                    // 128 x 384
static constexpr size_t OFF_MSG = OFF_INF + 128 * 384;  // 128 x 128
static constexpr size_t OFF_UPD = OFF_MSG + 128 * 128;  // 128 x 256
static constexpr size_t OFF_DEF = OFF_UPD + 128 * 256;  // 128 x 128
static constexpr size_t OFF_V1W = OFF_DEF + 128 * 128;  // 256 x 128
static constexpr size_t OFF_V2W = OFF_V1W + 256 * 128;  // 128 x 256
static constexpr size_t OFF_H1W = OFF_V2W + 128 * 256;  // 256 x 128
static constexpr size_t OFF_H2W = OFF_H1W + 256 * 128;  // 128 x 256
static constexpr size_t WTS_BYTES = 512 * 1024;         // weights region (padded)

extern "C" void kernel_launch(void* const* d_in, const int* in_sizes, int n_in,
                              void* d_out, int out_size, void* d_ws, size_t ws_size,
                              hipStream_t stream) {
  (void)in_sizes; (void)n_in; (void)out_size; (void)ws_size;
  const float* hexf  = (const float*)d_in[0];
  const float* vertf = (const float*)d_in[1];
  const float* W_inf = (const float*)d_in[2];  const float* b_inf = (const float*)d_in[3];
  const float* W_msg = (const float*)d_in[4];  const float* b_msg = (const float*)d_in[5];
  const float* W_upd = (const float*)d_in[6];  const float* b_upd = (const float*)d_in[7];
  const float* W_def = (const float*)d_in[8];  const float* b_def = (const float*)d_in[9];
  const float* vn_g  = (const float*)d_in[10]; const float* vn_b  = (const float*)d_in[11];
  const float* hn_g  = (const float*)d_in[12]; const float* hn_b  = (const float*)d_in[13];
  const float* Wv1   = (const float*)d_in[14]; const float* bv1   = (const float*)d_in[15];
  const float* Wv2   = (const float*)d_in[16]; const float* bv2   = (const float*)d_in[17];
  const float* Wh1   = (const float*)d_in[18]; const float* bh1   = (const float*)d_in[19];
  const float* Wh2   = (const float*)d_in[20]; const float* bh2   = (const float*)d_in[21];
  const int* vth  = (const int*)d_in[22];
  const int* htv  = (const int*)d_in[23];
  const int* vadj = (const int*)d_in[24];

  bf16_t* wts = (bf16_t*)d_ws;
  float* v1 = (float*)((char*)d_ws + WTS_BYTES);          // (B,N,128) f32
  float* hexout = (float*)d_out;                          // output 0: (B,T,128)
  float* vout = (float*)d_out + (size_t)B * T * H;        // output 1: (B,N,128)

  auto prep = [&](const float* W, bf16_t* Wt, int K, int Nout) {
    int n = K * Nout;
    k_prep<<<(n + 255) / 256, 256, 0, stream>>>(W, Wt, K, Nout);
  };
  prep(W_inf, wts + OFF_INF, 384, 128);
  prep(W_msg, wts + OFF_MSG, 128, 128);
  prep(W_upd, wts + OFF_UPD, 256, 128);
  prep(W_def, wts + OFF_DEF, 128, 128);
  prep(Wv1,   wts + OFF_V1W, 128, 256);
  prep(Wv2,   wts + OFF_V2W, 256, 128);
  prep(Wh1,   wts + OFF_H1W, 128, 256);
  prep(Wh2,   wts + OFF_H2W, 256, 128);

  k_inflate<<<(B * N) / 32, 256, 0, stream>>>(hexf, vertf, wts + OFF_INF, b_inf,
                                              vth, v1);
  k_msg<<<(B * N) / 32, 256, 0, stream>>>(v1, wts + OFF_MSG, b_msg,
                                          wts + OFF_UPD, b_upd, vn_g, vn_b,
                                          vadj, vout);
  k_mlp<<<(B * N) / 32, 256, 0, stream>>>(vout, wts + OFF_V1W, bv1,
                                          wts + OFF_V2W, bv2);
  k_deflate<<<(B * T) / 32, 256, 0, stream>>>(vout, hexf, wts + OFF_DEF, b_def,
                                              hn_g, hn_b, htv, hexout);
  k_mlp<<<(B * T) / 32, 256, 0, stream>>>(hexout, wts + OFF_H1W, bh1,
                                          wts + OFF_H2W, bh2);
}
